// SimpleLSTM_78804059947332
// MI455X (gfx1250) — compile-verified, matrix operand order checked by reading, not run on hardware
//
#include <hip/hip_runtime.h>
#include <hip/hip_bf16.h>

// ---------------------------------------------------------------------------
// Two stacked LSTMs (B=128, T=1024, FEAT=128, H=256) + dense(1) relu head.
// Strategy: per-timestep fused GEMM  g = [x_t | h_prev] @ [W;U]^T-packed
// using v_wmma_f32_16x16x32_bf16, then a pointwise gate kernel.
// Weights pre-packed to bf16 row-major-by-N so both A and B WMMA fragments
// load as two contiguous 16-byte chunks per lane.
// Per-step GEMM runs as 128 single-wave workgroups so the tiny latency-bound
// step spreads across ~128 WGPs instead of 32.
// ---------------------------------------------------------------------------

typedef __attribute__((ext_vector_type(16))) __bf16 v16bf;
typedef __attribute__((ext_vector_type(8)))  float  v8f;
typedef __attribute__((ext_vector_type(4)))  unsigned int u32x4;

#define B_SZ    128
#define T_SZ    1024
#define FEAT_SZ 128
#define H_SZ    256
#define G_SZ    1024   // 4*H

union V16 {
  u32x4 q[2];
  v16bf v;
};

__device__ __forceinline__ float fast_sigmoid(float x) {
  return 1.0f / (1.0f + __expf(-x));
}
__device__ __forceinline__ float fast_tanh(float x) {
  x = fminf(fmaxf(x, -15.0f), 15.0f);
  float e = __expf(2.0f * x);
  return (e - 1.0f) / (e + 1.0f);
}
__device__ __forceinline__ unsigned short f2bf(float f) {
  unsigned int u = __float_as_uint(f);
  u += 0x7fffu + ((u >> 16) & 1u);          // round-to-nearest-even
  return (unsigned short)(u >> 16);
}
__device__ __forceinline__ float bf2f(unsigned short h) {
  return __uint_as_float(((unsigned int)h) << 16);
}

// ---------------------------------------------------------------------------
// Per-step gate GEMM:  g[128,1024] = A[128,Kc] * Wt^T   (Wt is [1024][Kc] bf16)
// A row b = concat( xpart[b*xstride .. +xK] , hpart[b*hK .. +hK] ), Kc=xK+hK.
// Grid: 128 blocks x 32 thr (1 wave each). Each wave: 1 M-tile x 4 N-tiles.
// A fragment reused across the 4 WMMAs of each K-slice.
// ---------------------------------------------------------------------------
__global__ __launch_bounds__(32) void lstm_gate_gemm(
    const unsigned short* __restrict__ xpart, long xstride, int xK,
    const unsigned short* __restrict__ hpart,
    const unsigned short* __restrict__ Wt, int Kc,
    float* __restrict__ g)
{
  const int lane  = threadIdx.x & 31;
  const int gw    = blockIdx.x;                 // 0..127
  const int mtile = gw >> 4;                    // 0..7   (M = 128)
  const int n0    = (gw & 15) * 64;             // group of 4 N-tiles
  const int mrow  = lane & 15;                  // A: row, B: column
  const int kb    = (lane >> 4) * 8;            // K sub-chunk (0 or 8)
  const int bRow  = mtile * 16 + mrow;          // batch row
  const int hK    = Kc - xK;

  v8f acc0 = {}, acc1 = {}, acc2 = {}, acc3 = {};

  const unsigned short* wrow = Wt + (long)(n0 + mrow) * Kc + kb;

  for (int kk = 0; kk < Kc; kk += 32) {
    const unsigned short* asrc =
        (kk < xK) ? (xpart + (long)bRow * xstride + kk + kb)
                  : (hpart + (long)bRow * hK + (kk - xK) + kb);
    V16 a, b0, b1, b2, b3;
    a.q[0] = *(const u32x4*)(asrc);
    a.q[1] = *(const u32x4*)(asrc + 16);
    const unsigned short* bsrc = wrow + kk;
    b0.q[0] = *(const u32x4*)(bsrc);
    b0.q[1] = *(const u32x4*)(bsrc + 16);
    b1.q[0] = *(const u32x4*)(bsrc + 16l * Kc);
    b1.q[1] = *(const u32x4*)(bsrc + 16l * Kc + 16);
    b2.q[0] = *(const u32x4*)(bsrc + 32l * Kc);
    b2.q[1] = *(const u32x4*)(bsrc + 32l * Kc + 16);
    b3.q[0] = *(const u32x4*)(bsrc + 48l * Kc);
    b3.q[1] = *(const u32x4*)(bsrc + 48l * Kc + 16);

    acc0 = __builtin_amdgcn_wmma_f32_16x16x32_bf16(false, a.v, false, b0.v, (short)0, acc0, false, false);
    acc1 = __builtin_amdgcn_wmma_f32_16x16x32_bf16(false, a.v, false, b1.v, (short)0, acc1, false, false);
    acc2 = __builtin_amdgcn_wmma_f32_16x16x32_bf16(false, a.v, false, b2.v, (short)0, acc2, false, false);
    acc3 = __builtin_amdgcn_wmma_f32_16x16x32_bf16(false, a.v, false, b3.v, (short)0, acc3, false, false);
  }

  // C/D layout: VGPR r -> row (lane>=16 ? 8 : 0)+r, column lane&15
  float* gout = g + (long)(mtile * 16 + (lane >> 4) * 8) * G_SZ + n0 + (lane & 15);
  #pragma unroll
  for (int r = 0; r < 8; ++r) {
    gout[(long)r * G_SZ +  0] = acc0[r];
    gout[(long)r * G_SZ + 16] = acc1[r];
    gout[(long)r * G_SZ + 32] = acc2[r];
    gout[(long)r * G_SZ + 48] = acc3[r];
  }
}

// ---------------------------------------------------------------------------
// Pointwise gates + state update (Keras gate order i,f,c,o; bias added here).
// Writes next h as bf16 (ping-pong) and optionally appends to h1seq.
// ---------------------------------------------------------------------------
__global__ __launch_bounds__(256) void lstm_pointwise(
    const float* __restrict__ g, const float* __restrict__ bias,
    float* __restrict__ c, unsigned short* __restrict__ h_next,
    unsigned short* __restrict__ hseq, long hseq_row_stride)
{
  int idx = blockIdx.x * 256 + threadIdx.x;   // 0 .. B*H-1
  int b = idx >> 8;
  int j = idx & (H_SZ - 1);
  const float* gr = g + (long)b * G_SZ;
  float ig = fast_sigmoid(gr[j]           + bias[j]);
  float fg = fast_sigmoid(gr[H_SZ   + j]  + bias[H_SZ   + j]);
  float cd = fast_tanh   (gr[2*H_SZ + j]  + bias[2*H_SZ + j]);
  float og = fast_sigmoid(gr[3*H_SZ + j]  + bias[3*H_SZ + j]);
  float cn = fg * c[idx] + ig * cd;
  float h  = og * fast_tanh(cn);
  c[idx]      = cn;
  h_next[idx] = f2bf(h);
  if (hseq) hseq[(long)b * hseq_row_stride + j] = f2bf(h);
}

// ---------------------------------------------------------------------------
// One-time prep kernels
// ---------------------------------------------------------------------------
__global__ __launch_bounds__(256) void f32_to_bf16_kernel(
    const float* __restrict__ in, unsigned short* __restrict__ out, long n)
{
  long stride = (long)gridDim.x * 256;
  for (long i = (long)blockIdx.x * 256 + threadIdx.x; i < n; i += stride)
    out[i] = f2bf(in[i]);
}

// Wt[n][k] = (k < kW ? W[k][n] : U[k-kW][n]) as bf16;  W:[kW,1024] U:[Kc-kW,1024]
__global__ __launch_bounds__(256) void pack_weights_kernel(
    const float* __restrict__ W, const float* __restrict__ U,
    int kW, int Kc, unsigned short* __restrict__ Wt)
{
  long total  = (long)G_SZ * Kc;
  long stride = (long)gridDim.x * 256;
  for (long i = (long)blockIdx.x * 256 + threadIdx.x; i < total; i += stride) {
    int n = (int)(i / Kc);
    int k = (int)(i % Kc);
    float v = (k < kW) ? W[(long)k * G_SZ + n] : U[(long)(k - kW) * G_SZ + n];
    Wt[i] = f2bf(v);
  }
}

// out[b] = relu(h_last[b,:] . Wd + bd)
__global__ __launch_bounds__(128) void dense_relu_kernel(
    const unsigned short* __restrict__ h, const float* __restrict__ Wd,
    const float* __restrict__ bd, float* __restrict__ out)
{
  int b = threadIdx.x;
  float s = bd[0];
  #pragma unroll 8
  for (int j = 0; j < H_SZ; ++j) s += bf2f(h[b * H_SZ + j]) * Wd[j];
  out[b] = fmaxf(s, 0.0f);
}

// ---------------------------------------------------------------------------
extern "C" void kernel_launch(void* const* d_in, const int* in_sizes, int n_in,
                              void* d_out, int out_size, void* d_ws, size_t ws_size,
                              hipStream_t stream) {
  (void)in_sizes; (void)n_in; (void)out_size; (void)ws_size;
  const float* x  = (const float*)d_in[0];
  const float* W1 = (const float*)d_in[1];
  const float* U1 = (const float*)d_in[2];
  const float* b1 = (const float*)d_in[3];
  const float* W2 = (const float*)d_in[4];
  const float* U2 = (const float*)d_in[5];
  const float* b2 = (const float*)d_in[6];
  const float* Wd = (const float*)d_in[7];
  const float* bd = (const float*)d_in[8];

  char* ws = (char*)d_ws;
  size_t off = 0;
  auto alloc = [&](size_t bytes) -> void* {
    void* p = ws + off;
    off = (off + bytes + 255) & ~(size_t)255;
    return p;
  };
  unsigned short* xb    = (unsigned short*)alloc((size_t)B_SZ * T_SZ * FEAT_SZ * 2); // 32 MiB
  unsigned short* h1seq = (unsigned short*)alloc((size_t)B_SZ * T_SZ * H_SZ * 2);    // 64 MiB
  unsigned short* Wt1   = (unsigned short*)alloc((size_t)G_SZ * (FEAT_SZ + H_SZ) * 2);
  unsigned short* Wt2   = (unsigned short*)alloc((size_t)G_SZ * (2 * H_SZ) * 2);
  float*          g     = (float*)alloc((size_t)B_SZ * G_SZ * 4);                    // 512 KiB
  char* stateBase       = (char*)alloc(2 * 65536 + 131072 + 2 * 65536 + 131072);     // 512 KiB
  unsigned short* h1a = (unsigned short*)(stateBase);
  unsigned short* h1b = (unsigned short*)(stateBase + 65536);
  float*          c1  = (float*)(stateBase + 131072);
  unsigned short* h2a = (unsigned short*)(stateBase + 262144);
  unsigned short* h2b = (unsigned short*)(stateBase + 327680);
  float*          c2  = (float*)(stateBase + 393216);

  // zero h0/c0 for both layers (single contiguous memset, capture-safe)
  hipMemsetAsync(stateBase, 0, 524288, stream);

  // one-time prep
  f32_to_bf16_kernel<<<2048, 256, 0, stream>>>(x, xb, (long)B_SZ * T_SZ * FEAT_SZ);
  pack_weights_kernel<<<512, 256, 0, stream>>>(W1, U1, FEAT_SZ, FEAT_SZ + H_SZ, Wt1);
  pack_weights_kernel<<<512, 256, 0, stream>>>(W2, U2, H_SZ, 2 * H_SZ, Wt2);

  // Layer 1 recurrence (writes h1seq in bf16)
  unsigned short* hcur = h1a;
  unsigned short* hnxt = h1b;
  for (int t = 0; t < T_SZ; ++t) {
    lstm_gate_gemm<<<128, 32, 0, stream>>>(
        xb + (long)t * FEAT_SZ, (long)T_SZ * FEAT_SZ, FEAT_SZ,
        hcur, Wt1, FEAT_SZ + H_SZ, g);
    lstm_pointwise<<<(B_SZ * H_SZ) / 256, 256, 0, stream>>>(
        g, b1, c1, hnxt, h1seq + (long)t * H_SZ, (long)T_SZ * H_SZ);
    unsigned short* tmp = hcur; hcur = hnxt; hnxt = tmp;
  }

  // Layer 2 recurrence (keeps only final h)
  unsigned short* h2cur = h2a;
  unsigned short* h2nxt = h2b;
  for (int t = 0; t < T_SZ; ++t) {
    lstm_gate_gemm<<<128, 32, 0, stream>>>(
        h1seq + (long)t * H_SZ, (long)T_SZ * H_SZ, H_SZ,
        h2cur, Wt2, 2 * H_SZ, g);
    lstm_pointwise<<<(B_SZ * H_SZ) / 256, 256, 0, stream>>>(
        g, b2, c2, h2nxt, (unsigned short*)nullptr, 0);
    unsigned short* tmp = h2cur; h2cur = h2nxt; h2nxt = tmp;
  }

  // Dense(1) + relu
  dense_relu_kernel<<<1, 128, 0, stream>>>(h2cur, Wd, bd, (float*)d_out);
}
